// GNN_91001767067826
// MI455X (gfx1250) — matrix operand order repaired and checked by reading, not compiled, not run on hardware
//
#include <hip/hip_runtime.h>
#include <math.h>

#define NN 40000
#define NE 640000
#define NG 64

typedef __attribute__((ext_vector_type(16))) _Float16 v16h;
typedef __attribute__((ext_vector_type(2)))  _Float16 v2h;
typedef __attribute__((ext_vector_type(8)))  float    v8f;

union FragU { uint4 u4[2]; v16h v; };

// ---------------------------------------------------------------------------
// Edge-weight MLP: ew = sigmoid( relu(a*w1 + b1) @ w2 + b2 ), per edge.
// ---------------------------------------------------------------------------
__global__ __launch_bounds__(256) void k_edge_mlp(
    const float* __restrict__ ea, const float* __restrict__ w1,
    const float* __restrict__ b1, const float* __restrict__ w2,
    const float* __restrict__ b2, float* __restrict__ ew) {
  int e = blockIdx.x * 256 + threadIdx.x;
  if (e >= NE) return;
  float a = ea[e];
  float s = b2[0];
#pragma unroll 8
  for (int j = 0; j < 64; ++j) {
    float t = fmaxf(fmaf(a, w1[j], b1[j]), 0.f);
    s = fmaf(t, w2[j], s);
  }
  ew[e] = 1.f / (1.f + expf(-s));
}

__global__ __launch_bounds__(256) void k_fill(float* __restrict__ p, float v, int n) {
  int i = blockIdx.x * 256 + threadIdx.x;
  if (i < n) p[i] = v;
}

__global__ __launch_bounds__(256) void k_deg(const int* __restrict__ dst,
                                             const float* __restrict__ ew,
                                             float* __restrict__ deg) {
  int e = blockIdx.x * 256 + threadIdx.x;
  if (e < NE) atomicAdd(&deg[dst[e]], ew[e]);
}

__global__ __launch_bounds__(256) void k_dinv(float* __restrict__ deg_dinv,
                                              float* __restrict__ snorm) {
  int n = blockIdx.x * 256 + threadIdx.x;
  if (n >= NN) return;
  float d = deg_dinv[n];
  float di = (d > 0.f) ? rsqrtf(fmaxf(d, 1e-12f)) : 0.f;
  deg_dinv[n] = di;
  snorm[n] = di * di;
}

__global__ __launch_bounds__(256) void k_norm(const int* __restrict__ src,
                                              const int* __restrict__ dst,
                                              const float* __restrict__ ew,
                                              const float* __restrict__ dinv,
                                              float* __restrict__ nrm) {
  int e = blockIdx.x * 256 + threadIdx.x;
  if (e < NE) nrm[e] = dinv[src[e]] * ew[e] * dinv[dst[e]];
}

// ---------------------------------------------------------------------------
// Tiled WMMA GEMM: C[M,Nd] = act(A[M,K] @ W[K,Nd] + bias)
// Requires M%64==0, K%32==0, Nd%128==0. Block = 256 threads = 8 waves.
// Block tile 64x128. Wave w: wr=w&3 -> rows wr*16.., wc=w>>2 -> cols wc*64..
// Each wave: 4 f32 16x16 accumulators (one A fragment reused 4x).
//
// Software pipeline: double-buffered LDS; global loads for tile k+1 are
// batched into registers, WMMAs on tile k cover their latency, then a single
// convert+ds_store phase and one barrier per K-step.
//
// LDS layout is fragment-friendly (all fragment reads are 2x ds_load_b128):
//   As[buf][row][k]  row stride 80B (16B-multiple)  -> two 8-half runs
//   Bst[buf][col][k] transposed at staging time     -> two 8-half runs
// ---------------------------------------------------------------------------
__global__ __launch_bounds__(256) void k_gemm_wmma(
    const float* __restrict__ A, const float* __restrict__ W,
    const float* __restrict__ bias, float* __restrict__ C,
    int M, int K, int Nd, int relu_flag) {
  __shared__ __align__(16) _Float16 As[2][64][40];    // 2 x 64 rows x 32 k (+pad)
  __shared__ __align__(16) _Float16 Bst[2][128][40];  // 2 x 128 cols x 32 k (+pad)

  const int tid  = threadIdx.x;
  const int lane = tid & 31;
  const int wave = tid >> 5;
  const int wr = wave & 3, wc = wave >> 2;
  const int m0 = blockIdx.x * 64, n0 = blockIdx.y * 128;
  const int r = lane & 15, h = lane >> 4;

  v8f acc[4] = {v8f{}, v8f{}, v8f{}, v8f{}};

  float2 areg[4];
  float  brega[8], bregb[8];

  // Batched global loads for one K tile into registers (all loads issued
  // before any use -> single wait, full memory-level parallelism).
  auto load_tile = [&](int k0) {
#pragma unroll
    for (int i = 0; i < 4; ++i) {
      int idx = tid + i * 256;                 // 0..1023 A pairs
      int row = idx >> 4, k = (idx & 15) << 1;
      areg[i] = *(const float2*)&A[(size_t)(m0 + row) * K + (k0 + k)];
    }
#pragma unroll
    for (int i = 0; i < 8; ++i) {
      int idx = tid + i * 256;                 // 0..2047 B pairs
      int c = idx & 127, k = (idx >> 7) << 1;
      brega[i] = W[(size_t)(k0 + k) * Nd + (n0 + c)];
      bregb[i] = W[(size_t)(k0 + k + 1) * Nd + (n0 + c)];
    }
  };
  // Convert f32 -> packed f16 and store to the given LDS buffer.
  auto store_tile = [&](int buf) {
#pragma unroll
    for (int i = 0; i < 4; ++i) {
      int idx = tid + i * 256;
      int row = idx >> 4, k = (idx & 15) << 1;
      *(v2h*)&As[buf][row][k] = v2h{(_Float16)areg[i].x, (_Float16)areg[i].y};
    }
#pragma unroll
    for (int i = 0; i < 8; ++i) {
      int idx = tid + i * 256;
      int c = idx & 127, k = (idx >> 7) << 1;
      *(v2h*)&Bst[buf][c][k] = v2h{(_Float16)brega[i], (_Float16)bregb[i]};
    }
  };

  const int nk = K >> 5;
  load_tile(0);
  store_tile(0);
  __syncthreads();

  for (int kt = 0; kt < nk; ++kt) {
    const int  cur  = kt & 1;
    const bool more = (kt + 1) < nk;

    if (more) load_tile((kt + 1) << 5);        // loads in flight during WMMAs
    if (kt + 2 < nk) {                          // L2 prefetch two tiles ahead
      int kp = (kt + 2) << 5;
      __builtin_prefetch(&A[(size_t)(m0 + (tid >> 2)) * K + kp + ((tid & 3) << 3)], 0, 3);
      __builtin_prefetch(&W[(size_t)(kp + (tid >> 3)) * Nd + n0 + ((tid & 7) << 4)], 0, 3);
    }

    // ---- A fragment: 2x ds_load_b128 ----
    FragU fa;
    fa.u4[0] = *(const uint4*)&As[cur][wr * 16 + r][h * 8];
    fa.u4[1] = *(const uint4*)&As[cur][wr * 16 + r][16 + h * 8];

    // ---- 4 B fragments, 4 WMMAs ----
#pragma unroll
    for (int t = 0; t < 4; ++t) {
      FragU fb;
      const _Float16* bp = &Bst[cur][wc * 64 + t * 16 + r][h * 8];
      fb.u4[0] = *(const uint4*)(bp);
      fb.u4[1] = *(const uint4*)(bp + 16);
      acc[t] = __builtin_amdgcn_wmma_f32_16x16x32_f16(false, fa.v, false, fb.v,
                                                      (short)0, acc[t], false, false);
    }

    if (more) store_tile(cur ^ 1);             // fill the other buffer
    __syncthreads();                           // one barrier per K-step
  }

  // ---- epilogue: D(m,n): lane = (m>=8)*16 + n, vgpr = m&7 ----
  float bv[4];
#pragma unroll
  for (int t = 0; t < 4; ++t)
    bv[t] = bias ? bias[n0 + wc * 64 + t * 16 + r] : 0.f;

#pragma unroll
  for (int t = 0; t < 4; ++t) {
    int ncol = n0 + wc * 64 + t * 16 + r;
#pragma unroll
    for (int v = 0; v < 8; ++v) {
      int mrow = m0 + wr * 16 + h * 8 + v;
      float val = acc[t][v] + bv[t];
      if (relu_flag) val = fmaxf(val, 0.f);
      C[(size_t)mrow * Nd + ncol] = val;
    }
  }
}

// ---------------------------------------------------------------------------
// GCN aggregation: agg[n,:] = bias + lin[n,:]*snorm[n] (self loop), then
// per-edge atomic scatter of lin[src,:]*norm into agg[dst,:].
// ---------------------------------------------------------------------------
__global__ __launch_bounds__(256) void k_agg_init(const float* __restrict__ lin,
                                                  const float* __restrict__ bias,
                                                  const float* __restrict__ snorm,
                                                  float* __restrict__ agg) {
  int i = blockIdx.x * 256 + threadIdx.x;  // < NN*512
  int n = i >> 9, c = i & 511;
  agg[i] = bias[c] + lin[i] * snorm[n];
}

__global__ __launch_bounds__(256) void k_agg_edges(const float* __restrict__ lin,
                                                   const int* __restrict__ src,
                                                   const int* __restrict__ dst,
                                                   const float* __restrict__ nrm,
                                                   float* __restrict__ agg) {
  int wave = (blockIdx.x * 256 + threadIdx.x) >> 5;  // one wave32 per edge
  int lane = threadIdx.x & 31;
  if (wave >= NE) return;
  int s = src[wave], d = dst[wave];
  float nv = nrm[wave];
  const float4* hs = (const float4*)(lin + (size_t)s * 512);
  float* ad = agg + (size_t)d * 512;
#pragma unroll
  for (int it = 0; it < 4; ++it) {
    int q = lane + it * 32;            // float4 index, 0..127
    float4 v = hs[q];
    int c = q << 2;
    atomicAdd(ad + c + 0, v.x * nv);
    atomicAdd(ad + c + 1, v.y * nv);
    atomicAdd(ad + c + 2, v.z * nv);
    atomicAdd(ad + c + 3, v.w * nv);
  }
}

// dst = relu(src) (add_flag==0) or dst += relu(src) (add_flag==1)
__global__ __launch_bounds__(256) void k_post(const float* __restrict__ src,
                                              float* __restrict__ dst,
                                              int add_flag, int n) {
  int i = blockIdx.x * 256 + threadIdx.x;
  if (i >= n) return;
  float r = fmaxf(src[i], 0.f);
  dst[i] = add_flag ? (dst[i] + r) : r;
}

// ---------------------------------------------------------------------------
// Global mean pool + head
// ---------------------------------------------------------------------------
__global__ __launch_bounds__(256) void k_pool_cnt(const int* __restrict__ batch,
                                                  float* __restrict__ cnt) {
  int n = blockIdx.x * 256 + threadIdx.x;
  if (n < NN) atomicAdd(&cnt[batch[n]], 1.f);
}

__global__ __launch_bounds__(256) void k_pool_accum(const float* __restrict__ skip,
                                                    const int* __restrict__ batch,
                                                    float* __restrict__ pooled) {
  int i = blockIdx.x * 256 + threadIdx.x;  // < NN*128
  int n = i >> 7, q = (i & 127) << 2;
  int b = batch[n];
  float4 v = *(const float4*)(skip + (size_t)n * 512 + q);
  float* pp = pooled + (size_t)b * 512 + q;
  atomicAdd(pp + 0, v.x);
  atomicAdd(pp + 1, v.y);
  atomicAdd(pp + 2, v.z);
  atomicAdd(pp + 3, v.w);
}

__global__ __launch_bounds__(256) void k_pool_div(float* __restrict__ pooled,
                                                  const float* __restrict__ cnt) {
  int i = blockIdx.x * 256 + threadIdx.x;
  if (i < NG * 512) pooled[i] /= fmaxf(cnt[i >> 9], 1.f);
}

__global__ __launch_bounds__(256) void k_head(const float* __restrict__ fc1o,
                                              const float* __restrict__ hw,
                                              const float* __restrict__ hb,
                                              float* __restrict__ out) {
  int i = blockIdx.x * 256 + threadIdx.x;
  if (i >= NG * 10) return;
  int g = i / 10, c = i - g * 10;
  float s = hb[c];
  const float* f = fc1o + (size_t)g * 1024;
  for (int k = 0; k < 1024; ++k) s = fmaf(f[k], hw[k * 10 + c], s);
  out[i] = s;
}

// ---------------------------------------------------------------------------
extern "C" void kernel_launch(void* const* d_in, const int* in_sizes, int n_in,
                              void* d_out, int out_size, void* d_ws, size_t ws_size,
                              hipStream_t stream) {
  const float* x      = (const float*)d_in[0];
  const int*   ei     = (const int*)d_in[1];
  const float* eattr  = (const float*)d_in[2];
  const int*   batch  = (const int*)d_in[3];
  const float* emb_w1 = (const float*)d_in[4];
  const float* emb_b1 = (const float*)d_in[5];
  const float* emb_w2 = (const float*)d_in[6];
  const float* emb_b2 = (const float*)d_in[7];
  const float* ep_w1  = (const float*)d_in[8];
  const float* ep_b1  = (const float*)d_in[9];
  const float* ep_w2  = (const float*)d_in[10];
  const float* ep_b2  = (const float*)d_in[11];
  const float* wc[6]  = {(const float*)d_in[12], (const float*)d_in[14],
                         (const float*)d_in[16], (const float*)d_in[18],
                         (const float*)d_in[20], (const float*)d_in[22]};
  const float* bc[6]  = {(const float*)d_in[13], (const float*)d_in[15],
                         (const float*)d_in[17], (const float*)d_in[19],
                         (const float*)d_in[21], (const float*)d_in[23]};
  const float* fc1_w  = (const float*)d_in[24];
  const float* fc1_b  = (const float*)d_in[25];
  const float* head_w = (const float*)d_in[26];
  const float* head_b = (const float*)d_in[27];
  float* out = (float*)d_out;

  // Workspace carve (floats): ~292 MB total.
  float* ws     = (float*)d_ws;
  float* ew     = ws;  ws += NE;
  float* dinv   = ws;  ws += NN;            // holds deg, then dinv in place
  float* snorm  = ws;  ws += NN;
  float* nrm    = ws;  ws += NE;
  float* h0     = ws;  ws += (size_t)NN * 256;
  float* lin    = ws;  ws += (size_t)NN * 512;   // GEMM out; also fc1 out
  float* agg    = ws;  ws += (size_t)NN * 512;   // aggregation / activation
  float* skip   = ws;  ws += (size_t)NN * 512;
  float* pooled = ws;  ws += NG * 512;
  float* cnt    = ws;  ws += NG;                 // contiguous after pooled

  const int* srcv = ei;
  const int* dstv = ei + NE;

  const int EB  = (NE + 255) / 256;         // edge-count blocks
  const int NB  = (NN + 255) / 256;         // node-count blocks
  const int NDB = (NN * 512) / 256;         // node*512 element blocks (80000)
  const int EWB = (NE * 32) / 256;          // one wave per edge (80000)

  // --- normalization pipeline ---
  k_edge_mlp<<<EB, 256, 0, stream>>>(eattr, ep_w1, ep_b1, ep_w2, ep_b2, ew);
  k_fill<<<NB, 256, 0, stream>>>(dinv, 1.0f, NN);        // deg = self-loop weight
  k_deg<<<EB, 256, 0, stream>>>(dstv, ew, dinv);
  k_dinv<<<NB, 256, 0, stream>>>(dinv, snorm);
  k_norm<<<EB, 256, 0, stream>>>(srcv, dstv, ew, dinv, nrm);

  // --- embedding MLP (WMMA) ---
  k_gemm_wmma<<<dim3(NN / 64, 2), 256, 0, stream>>>(x,   emb_w1, emb_b1, lin, NN, 32,  256, 1);
  k_gemm_wmma<<<dim3(NN / 64, 2), 256, 0, stream>>>(lin, emb_w2, emb_b2, h0,  NN, 256, 256, 1);

  // --- 6 GCN conv layers ---
  auto conv = [&](const float* in, int Kd, const float* W, const float* B,
                  float* post_dst, int add_flag) {
    k_gemm_wmma<<<dim3(NN / 64, 4), 256, 0, stream>>>(in, W, nullptr, lin, NN, Kd, 512, 0);
    k_agg_init<<<NDB, 256, 0, stream>>>(lin, B, snorm, agg);
    k_agg_edges<<<EWB, 256, 0, stream>>>(lin, srcv, dstv, nrm, agg);
    k_post<<<NDB, 256, 0, stream>>>(agg, post_dst, add_flag, NN * 512);
  };
  conv(h0,   256, wc[0], bc[0], skip, 0);   // skip  = relu(h1)
  conv(skip, 512, wc[1], bc[1], agg,  0);   // agg   = relu(h2)
  conv(agg,  512, wc[2], bc[2], skip, 1);   // skip += relu(h3)
  conv(skip, 512, wc[3], bc[3], agg,  0);   // agg   = relu(h4)
  conv(agg,  512, wc[4], bc[4], agg,  0);   // agg   = relu(h5)
  conv(agg,  512, wc[5], bc[5], skip, 1);   // skip += relu(h6)

  // --- global mean pool ---
  k_fill<<<(NG * 512 + NG + 255) / 256, 256, 0, stream>>>(pooled, 0.f, NG * 512 + NG);
  k_pool_cnt<<<NB, 256, 0, stream>>>(batch, cnt);
  k_pool_accum<<<(NN * 128) / 256, 256, 0, stream>>>(skip, batch, pooled);
  k_pool_div<<<(NG * 512) / 256, 256, 0, stream>>>(pooled, cnt);

  // --- head: fc1 (WMMA) + 10-class linear ---
  k_gemm_wmma<<<dim3(1, 8), 256, 0, stream>>>(pooled, fc1_w, fc1_b, lin, 64, 512, 1024, 1);
  k_head<<<(NG * 10 + 255) / 256, 256, 0, stream>>>(lin, head_w, head_b, out);

  (void)in_sizes; (void)n_in; (void)out_size; (void)ws_size;
}
